// VectorQuantizer_43791486550285
// MI455X (gfx1250) — matrix operand order repaired and checked by reading, not compiled, not run on hardware
//
#include <hip/hip_runtime.h>

// ---------- types for WMMA ----------
typedef __bf16 bf16_t;
typedef __attribute__((ext_vector_type(16))) __bf16 v16bf;
typedef __attribute__((ext_vector_type(8)))  __bf16 v8bf;
typedef __attribute__((ext_vector_type(8)))  float  v8f;

union V16 { v16bf v; v8bf h[2]; };

#define N_EMB     512
#define EMB       64
#define CB_STRIDE 72            // halfs; 144B rows -> 16B aligned, spreads banks
#define MROWS     128           // pixels per block
#define ZSTRIDE   72            // halfs
#define NPIX      (32 * 64 * 64)        // 131072 pixel vectors
#define TOTAL_ELEMS (NPIX * EMB)        // 8388608 elements of z / z_q

__global__ void vq_init_out(float* out) {
    if (threadIdx.x == 0 && blockIdx.x == 0) out[0] = 0.0f;
}

// Branchless xor-shuffle argmin step via ds_swizzle (group-of-32: bit15=0,
// xor_mask=offset[14:10], or_mask=0, and_mask=0x1f). All lanes active.
template <int XMASK>
__device__ __forceinline__ void argmin_step(float& v, int& i) {
    constexpr int offs = (XMASK << 10) | 0x1f;
    float ov = __int_as_float(__builtin_amdgcn_ds_swizzle(__float_as_int(v), offs));
    int   oi = __builtin_amdgcn_ds_swizzle(i, offs);
    bool take = (ov < v) || (ov == v && oi < i);   // first-hit argmin semantics
    v = take ? ov : v;
    i = take ? oi : i;
}

__global__ __launch_bounds__(256) void vq_kernel(const float* __restrict__ z,
                                                 const float* __restrict__ cb,
                                                 float* __restrict__ out) {
    __shared__ bf16_t s_cb[N_EMB * CB_STRIDE];   // -2*e, bf16
    __shared__ float  s_enorm[N_EMB];            // ||e||^2, fp32
    __shared__ bf16_t s_z[MROWS * ZSTRIDE];      // z tile, bf16
    __shared__ int    s_idx[MROWS];              // argmin per row
    __shared__ float  s_loss[8];

    const int tid  = threadIdx.x;
    const int lane = tid & 31;
    const int wave = tid >> 5;
    const int n    = lane & 15;   // column-in-tile / row-in-tile selector
    const int hi   = lane >> 4;   // half-wave

    const int pix0 = blockIdx.x * MROWS;    // 128 consecutive pixels, same image
    const int b    = pix0 >> 12;            // batch index (4096 pixels per image)
    const int p0   = pix0 & 4095;           // flat (h*W + w) base
    const float* zbase = z + (size_t)b * EMB * 4096 + p0;

    // ---- stage codebook: bf16(-2e) + fp32 norms ----
    for (int c = tid; c < N_EMB; c += 256) {
        const float* row = cb + c * EMB;
        bf16_t* dst = s_cb + c * CB_STRIDE;
        float acc = 0.0f;
        #pragma unroll
        for (int k = 0; k < EMB; ++k) {
            float v = row[k];
            acc += v * v;
            dst[k] = (bf16_t)(-2.0f * v);
        }
        s_enorm[c] = acc;
    }
    // ---- stage z tile (coalesced: contiguous in pixel per channel) ----
    for (int i = tid; i < MROWS * EMB; i += 256) {
        int c = i >> 7, pix = i & 127;
        s_z[pix * ZSTRIDE + c] = (bf16_t)zbase[(size_t)c * 4096 + pix];
    }
    __syncthreads();

    // ---- A operand: wave's 16 rows, K=64 as two K=32 WMMA operands ----
    // 16-bit A 16x32 layout: lanes0-15 M=0..15 {K0..7,K16..23}; lanes16-31 {K8..15,K24..31}
    const bf16_t* arow = s_z + (wave * 16 + n) * ZSTRIDE;
    V16 A0, A1;
    A0.h[0] = *(const v8bf*)(arow + 0  + hi * 8);
    A0.h[1] = *(const v8bf*)(arow + 16 + hi * 8);
    A1.h[0] = *(const v8bf*)(arow + 32 + hi * 8);
    A1.h[1] = *(const v8bf*)(arow + 48 + hi * 8);

    float best[8];
    int   bidx[8];
    #pragma unroll
    for (int r = 0; r < 8; ++r) { best[r] = 3.4e38f; bidx[r] = 0; }

    // ---- sweep 512 codes in 32 column tiles of 16 (unroll 2: overlap the
    //      WMMA->VALU hazard of tile i with B-loads/WMMAs of tile i+1) ----
    #pragma unroll 2
    for (int ct = 0; ct < 32; ++ct) {
        int code = ct * 16 + n;
        // B 32x16 layout: col = lane&15; lanes0-15 {K0..7,K16..23}; lanes16-31 {K8..15,K24..31}
        const bf16_t* brow = s_cb + code * CB_STRIDE;
        V16 B0, B1;
        B0.h[0] = *(const v8bf*)(brow + 0  + hi * 8);
        B0.h[1] = *(const v8bf*)(brow + 16 + hi * 8);
        B1.h[0] = *(const v8bf*)(brow + 32 + hi * 8);
        B1.h[1] = *(const v8bf*)(brow + 48 + hi * 8);

        float en = s_enorm[code];
        v8f acc = {en, en, en, en, en, en, en, en};   // D = ||e||^2 - 2 z.e
        acc = __builtin_amdgcn_wmma_f32_16x16x32_bf16(false, A0.v, false, B0.v,
                                                      (short)0, acc, false, false);
        acc = __builtin_amdgcn_wmma_f32_16x16x32_bf16(false, A1.v, false, B1.v,
                                                      (short)0, acc, false, false);
        #pragma unroll
        for (int r = 0; r < 8; ++r) {
            if (acc[r] < best[r]) { best[r] = acc[r]; bidx[r] = code; }
        }
    }

    // ---- argmin across the 16 columns held by each half-wave (branchless) ----
    #pragma unroll
    for (int r = 0; r < 8; ++r) {
        argmin_step<1>(best[r], bidx[r]);
        argmin_step<2>(best[r], bidx[r]);
        argmin_step<4>(best[r], bidx[r]);
        argmin_step<8>(best[r], bidx[r]);
    }
    // lanes0-15 slot r -> row r ; lanes16-31 slot r -> row r+8
    #pragma unroll
    for (int r = 0; r < 8; ++r)
        if (n == r) s_idx[wave * 16 + r + hi * 8] = bidx[r];
    __syncthreads();

    // ---- output: gather codes (fp32, L2-resident), write z_q [B,C,H,W], loss ----
    float* zq = out + 1;
    float lsum = 0.0f;
    for (int i = tid; i < MROWS * EMB; i += 256) {
        int c = i >> 7, pix = i & 127;
        int code = s_idx[pix];
        float e  = cb[code * EMB + c];
        float zv = zbase[(size_t)c * 4096 + pix];
        float d  = e - zv;
        lsum += d * d;
        zq[(size_t)b * EMB * 4096 + (size_t)c * 4096 + p0 + pix] = e;
    }
    if (tid < MROWS)
        out[1 + (size_t)TOTAL_ELEMS + pix0 + tid] = (float)s_idx[tid];

    // block loss reduction -> atomic
    #pragma unroll
    for (int mask = 16; mask >= 1; mask >>= 1) lsum += __shfl_xor(lsum, mask, 32);
    if (lane == 0) s_loss[wave] = lsum;
    __syncthreads();
    if (tid == 0) {
        float t = 0.0f;
        #pragma unroll
        for (int w = 0; w < 8; ++w) t += s_loss[w];
        atomicAdd(out, t * (1.0f / (float)TOTAL_ELEMS));
    }
}

extern "C" void kernel_launch(void* const* d_in, const int* in_sizes, int n_in,
                              void* d_out, int out_size, void* d_ws, size_t ws_size,
                              hipStream_t stream) {
    const float* z  = (const float*)d_in[0];   // [32,64,64,64] fp32
    const float* cb = (const float*)d_in[1];   // [512,64] fp32
    float* out = (float*)d_out;                // [1 + 8388608 + 131072] fp32
    vq_init_out<<<dim3(1), dim3(1), 0, stream>>>(out);
    vq_kernel<<<dim3(NPIX / MROWS), dim3(256), 0, stream>>>(z, cb, out);
}
